// EdgeConv_936302871069
// MI455X (gfx1250) — matrix operand order repaired and checked by reading, not compiled, not run on hardware
//
#include <hip/hip_runtime.h>
#include <stdint.h>

#define N_NODES 50000
#define N_EDGES 600000
#define D 128   // D_IN == D_OUT == 128

typedef __attribute__((ext_vector_type(16))) _Float16 v16h;
typedef __attribute__((ext_vector_type(8)))  _Float16 v8h;
typedef __attribute__((ext_vector_type(8)))  float    v8f;
typedef __attribute__((ext_vector_type(4)))  unsigned int v4u;

union Frag16 { v16h h; v4u u[2]; };

// ---------------------------------------------------------------------------
// Weight pre-swizzle into this code's B-fragment order (16-bit B, 32x16 per fragment, contiguous K as in A):
//   lane = column N within 16-wide tile (lanes 16..31 repeat N, upper K half)
//   element j (0..15): K = kc*32 + (lane>=16 ? 16 : 0) + j
// Fragment order: (kc, nt) major, then lane, then j  -> 16-byte contiguous.
// ---------------------------------------------------------------------------

// Wc = [W1top - W1bot | W1bot]  (128 x 256), f16, swizzled. 4 kc * 16 nt frags.
__global__ void prep_w1_swz(const float* __restrict__ W1, _Float16* __restrict__ Wc) {
    int tid = blockIdx.x * blockDim.x + threadIdx.x;     // 4*16*32*16 = 32768
    if (tid >= 4 * 16 * 32 * 16) return;
    int j    = tid & 15;
    int lane = (tid >> 4) & 31;
    int nt   = (tid >> 9) & 15;
    int kc   = tid >> 13;
    int k = kc * 32 + ((lane >> 4) << 4) + j;
    int n = nt * 16 + (lane & 15);
    float v;
    if (n < D) v = W1[k * D + n] - W1[(k + D) * D + n];   // Wa = W1top - W1bot
    else       v = W1[(k + D) * D + (n - D)];             // Wb = W1bot
    Wc[tid] = (_Float16)v;
}

// W2 (128 x 128), f16, swizzled. 4 kc * 8 nt frags.
__global__ void prep_w2_swz(const float* __restrict__ W2, _Float16* __restrict__ W2s) {
    int tid = blockIdx.x * blockDim.x + threadIdx.x;     // 4*8*32*16 = 16384
    if (tid >= 4 * 8 * 32 * 16) return;
    int j    = tid & 15;
    int lane = (tid >> 4) & 31;
    int nt   = (tid >> 9) & 7;
    int kc   = tid >> 12;
    int k = kc * 32 + ((lane >> 4) << 4) + j;
    int n = nt * 16 + (lane & 15);
    W2s[tid] = (_Float16)W2[k * D + n];
}

__global__ void init_out(float* __restrict__ out, int n) {
    int i = blockIdx.x * blockDim.x + threadIdx.x;
    if (i < n) out[i] = -__builtin_inff();
}

__global__ void finalize_out(float* __restrict__ out, int n) {
    int i = blockIdx.x * blockDim.x + threadIdx.x;
    if (i < n) {
        float v = out[i];
        out[i] = (v == -__builtin_inff()) ? 0.0f : v;
    }
}

// ---------------------------------------------------------------------------
// Node projection: P[n][0:256] = x[n][:] @ [Wa | Wb], f16 output.
// One wave per 16-node tile; 8 waves / block. A staged f16 in LDS, then the
// 4 A-fragments are hoisted into registers and reused across all 16 N-tiles.
// ---------------------------------------------------------------------------
__global__ void __launch_bounds__(256)
node_proj_wmma(const float* __restrict__ x, const _Float16* __restrict__ Wc,
               _Float16* __restrict__ P) {
    __shared__ __align__(16) _Float16 lds_a[8][16 * D];   // 32 KB staging

    int tidb = threadIdx.x;
    int wave = tidb >> 5;
    int lane = tidb & 31;
    int tile = blockIdx.x * 8 + wave;
    bool active = tile < (N_NODES / 16);

    if (active) {
        // lane covers row m = lane/2, halves ks..ks+63
        int m  = lane >> 1;
        int ks = (lane & 1) * 64;
        long long nrow = (long long)tile * 16 + m;
        const float* xr = x + nrow * D + ks;
        _Float16* dst = &lds_a[wave][m * D + ks];
        #pragma unroll
        for (int j = 0; j < 64; j += 4) {
            float4 f = *(const float4*)(xr + j);
            dst[j + 0] = (_Float16)f.x;
            dst[j + 1] = (_Float16)f.y;
            dst[j + 2] = (_Float16)f.z;
            dst[j + 3] = (_Float16)f.w;
        }
    }
    __syncthreads();
    if (!active) return;

    int mA  = lane & 15;
    int kbA = ((lane >> 4) & 1) * 16;         // A layout upper-half K offset
    const v4u* ap  = (const v4u*)&lds_a[wave][0];
    const v4u* wcp = (const v4u*)Wc;           // 64 KB, L2/WGP$ resident
    int mbase = ((lane >> 4) & 1) * 8;
    int ncol  = lane & 15;

    // Load all 4 A fragments once; reuse across the 16 N-tiles.
    Frag16 a[4];
    #pragma unroll
    for (int kc = 0; kc < 4; ++kc) {
        int abase = (mA * D + kc * 32 + kbA) >> 3;         // v4u index
        a[kc].u[0] = ap[abase];
        a[kc].u[1] = ap[abase + 1];                        // +8 halves
    }

    for (int nt = 0; nt < 16; ++nt) {
        v8f acc = {};
        #pragma unroll
        for (int kc = 0; kc < 4; ++kc) {
            Frag16 b;
            int bbase = ((kc * 16 + nt) * 32 + lane) * 2;
            b.u[0] = wcp[bbase];
            b.u[1] = wcp[bbase + 1];
            acc = __builtin_amdgcn_wmma_f32_16x16x32_f16(
                false, a[kc].h, false, b.h, (short)0, acc, false, false);
        }
        #pragma unroll
        for (int v = 0; v < 8; ++v) {
            long long node = (long long)tile * 16 + mbase + v;
            P[node * 256 + nt * 16 + ncol] = (_Float16)acc[v];
        }
    }
}

// ---------------------------------------------------------------------------
// Edge kernel: h = relu(Pa[col] + Pb[row] + b1); msg = h @ W2 + b2;
// out[col] = max(out[col], msg) via native f32 atomic max.
// One wave per 16-edge tile; 8 waves / block. W2 fragments cached in LDS.
// A fragments and column offsets hoisted out of the N-tile loop.
// ---------------------------------------------------------------------------
__global__ void __launch_bounds__(256)
edge_wmma(const _Float16* __restrict__ P, const long long* __restrict__ eidx,
          const float* __restrict__ b1, const _Float16* __restrict__ W2s,
          const float* __restrict__ b2, float* __restrict__ out) {
    __shared__ __align__(16) _Float16 lds_w[4 * 8 * 32 * 16]; // 32 KB W2 swizzled
    __shared__ __align__(16) _Float16 lds_h[8][16 * D];       // 32 KB staging

    int tidb = threadIdx.x;
    {   // cooperative load of swizzled W2 into LDS
        const v4u* src = (const v4u*)W2s;
        v4u* dst = (v4u*)lds_w;
        #pragma unroll
        for (int i = tidb; i < (4 * 8 * 32 * 16) / 8; i += 256) dst[i] = src[i];
    }

    int wave = tidb >> 5;
    int lane = tidb & 31;
    int tile = blockIdx.x * 8 + wave;
    bool active = tile < (N_EDGES / 16);

    if (active) {
        // Stage h = relu(Pa[col] + Pb[row] + b1) as f16; lane covers half a row.
        int m  = lane >> 1;
        int ks = (lane & 1) * 64;
        int e  = tile * 16 + m;
        long long row = eidx[e];
        long long col = eidx[N_EDGES + e];
        const _Float16* pa = P + col * 256 + ks;        // cols 0..127   = Pa
        const _Float16* pb = P + row * 256 + 128 + ks;  // cols 128..255 = Pb
        _Float16* dst = &lds_h[wave][m * D + ks];
        #pragma unroll
        for (int j = 0; j < 64; j += 8) {
            v8h a = *(const v8h*)(pa + j);
            v8h b = *(const v8h*)(pb + j);
            #pragma unroll
            for (int t = 0; t < 8; ++t) {
                float v = (float)a[t] + (float)b[t] + b1[ks + j + t];
                dst[j + t] = (_Float16)(v > 0.0f ? v : 0.0f);
            }
        }
    }
    __syncthreads();
    if (!active) return;

    int mA  = lane & 15;
    int kbA = ((lane >> 4) & 1) * 16;
    const v4u* hp = (const v4u*)&lds_h[wave][0];
    const v4u* wp = (const v4u*)lds_w;
    int mbase = ((lane >> 4) & 1) * 8;
    int ncol  = lane & 15;

    // Load all 4 A fragments once; reuse across the 8 N-tiles.
    Frag16 a[4];
    #pragma unroll
    for (int kc = 0; kc < 4; ++kc) {
        int abase = (mA * D + kc * 32 + kbA) >> 3;
        a[kc].u[0] = hp[abase];
        a[kc].u[1] = hp[abase + 1];
    }

    // Hoist the 8 destination-row offsets (nt-invariant) out of the loop so
    // the atomics' "memory" clobber can't force eidx re-loads.
    int coff[8];
    #pragma unroll
    for (int v = 0; v < 8; ++v) {
        int e = tile * 16 + mbase + v;
        coff[v] = (int)eidx[N_EDGES + e] * D;   // col * 128 < 2^31
    }

    for (int nt = 0; nt < 8; ++nt) {
        v8f acc = {};
        #pragma unroll
        for (int kc = 0; kc < 4; ++kc) {
            Frag16 b;
            int bbase = ((kc * 8 + nt) * 32 + lane) * 2;
            b.u[0] = wp[bbase];
            b.u[1] = wp[bbase + 1];
            acc = __builtin_amdgcn_wmma_f32_16x16x32_f16(
                false, a[kc].h, false, b.h, (short)0, acc, false, false);
        }
        float bias = b2[nt * 16 + ncol];
        #pragma unroll
        for (int v = 0; v < 8; ++v) {
            float val = acc[v] + bias;
            float* p = out + coff[v] + nt * 16 + ncol;
            // Native CDNA5 float atomic max, device scope, no return.
            asm volatile("global_atomic_max_num_f32 %0, %1, off scope:SCOPE_DEV"
                         :: "v"((uint64_t)(uintptr_t)p), "v"(val)
                         : "memory");
        }
    }
}

// ---------------------------------------------------------------------------
extern "C" void kernel_launch(void* const* d_in, const int* in_sizes, int n_in,
                              void* d_out, int out_size, void* d_ws, size_t ws_size,
                              hipStream_t stream) {
    const float*     x    = (const float*)d_in[0];
    const long long* eidx = (const long long*)d_in[1];   // int64 edge_index [2, E]
    const float*     W1   = (const float*)d_in[2];
    const float*     b1   = (const float*)d_in[3];
    const float*     W2   = (const float*)d_in[4];
    const float*     b2   = (const float*)d_in[5];
    float* out = (float*)d_out;

    // Workspace layout: [Wc_swz 64KB][W2_swz 32KB][P f16 50000*256 = 25.6MB]
    char* ws = (char*)d_ws;
    _Float16* Wc  = (_Float16*)ws;
    _Float16* W2s = (_Float16*)(ws + 65536);
    _Float16* P   = (_Float16*)(ws + 65536 + 32768);

    prep_w1_swz<<<32768 / 256, 256, 0, stream>>>(W1, Wc);
    prep_w2_swz<<<16384 / 256, 256, 0, stream>>>(W2, W2s);
    init_out<<<(N_NODES * D + 255) / 256, 256, 0, stream>>>(out, N_NODES * D);

    node_proj_wmma<<<((N_NODES / 16) + 7) / 8, 256, 0, stream>>>(x, Wc, P);
    edge_wmma<<<((N_EDGES / 16) + 7) / 8, 256, 0, stream>>>(P, eidx, b1, W2s, b2, out);

    finalize_out<<<(N_NODES * D + 255) / 256, 256, 0, stream>>>(out, N_NODES * D);
}
